// GNN_Model_59665685676076
// MI455X (gfx1250) — matrix operand order repaired and checked by reading, not compiled, not run on hardware
//
#include <hip/hip_runtime.h>
#include <hip/hip_bf16.h>

// ---------------------------------------------------------------------------
// CDNA5 (gfx1250) fused GNN scorer.
//   K0: convert feats fp32 -> bf16 table in workspace (128 MB, fits 192MB L2;
//       makes the irregular gather L2-resident and halves gathered bytes)
//   K1: pack fp32 weights -> bf16 WMMA B-fragment order (per launch, cheap)
//   K2: per-query neighbor gather (async global->LDS b128) + LSTM gates via
//       v_wmma_f32_16x16x32_bf16 + cell-state mean -> c[B,128]
//   K3: fused mv_aggregate (Wo gate, Wmv proj) for query & mv + final MLP
// Activations use CDNA5 hardware transcendentals (v_tanh/v_rcp/v_exp).
// ---------------------------------------------------------------------------

typedef __attribute__((ext_vector_type(16))) __bf16          v16bf;
typedef __attribute__((ext_vector_type(8)))  float           v8f;
typedef __attribute__((ext_vector_type(8)))  unsigned short  ush8;

// exact pointee type expected by the async-LDS builtins (per hipcc diagnostic)
typedef int v4i __attribute__((__vector_size__(16)));
typedef __attribute__((address_space(1))) v4i* gptr_v4i;   // global
typedef __attribute__((address_space(3))) v4i* lptr_v4i;   // LDS

#define DD  128
#define KNN 32

#if __has_builtin(__builtin_amdgcn_global_load_async_to_lds_b128)
#define USE_ASYNC_GATHER 1
#else
#define USE_ASYNC_GATHER 0
#endif

__device__ __forceinline__ void wait_async0() {
#if __has_builtin(__builtin_amdgcn_s_wait_asynccnt)
    __builtin_amdgcn_s_wait_asynccnt(0);
#else
    asm volatile("s_wait_asynccnt 0" ::: "memory");
#endif
}

// ---- scalar bf16 helpers (RNE) --------------------------------------------
__device__ __forceinline__ unsigned short f2bf(float f) {
    unsigned int u = __float_as_uint(f);
    unsigned int r = u + 0x7FFFu + ((u >> 16) & 1u);
    return (unsigned short)(r >> 16);
}
__device__ __forceinline__ float bf2f(unsigned short h) {
    return __uint_as_float(((unsigned int)h) << 16);
}

// pack two floats -> two bf16 in one dword (v_cvt_pk_bf16_f32 when available)
#if __has_builtin(__builtin_amdgcn_cvt_pk_bf16_f32)
__device__ __forceinline__ unsigned int pk2bf(float lo, float hi) {
    auto r = __builtin_amdgcn_cvt_pk_bf16_f32(lo, hi);
    unsigned int u;
    __builtin_memcpy(&u, &r, 4);
    return u;
}
#else
__device__ __forceinline__ unsigned int pk2bf(float lo, float hi) {
    return ((unsigned int)f2bf(hi) << 16) | (unsigned int)f2bf(lo);
}
#endif

// ---- fast activations on CDNA5 trans pipe ---------------------------------
__device__ __forceinline__ float fast_sigmoid(float x) {
    return __builtin_amdgcn_rcpf(1.0f + __expf(-x));   // v_exp + v_add + v_rcp
}
__device__ __forceinline__ float fast_tanh(float x) {
#if __has_builtin(__builtin_amdgcn_tanhf)
    return __builtin_amdgcn_tanhf(x);                  // v_tanh_f32 (gfx1250)
#else
    float e = __expf(2.0f * x);
    return 1.0f - 2.0f * __builtin_amdgcn_rcpf(e + 1.0f);
#endif
}

union FragU { v16bf v; ush8 h[2]; };

// A fragment (16x32 bf16, row-major source with row stride `stride` u16 elems).
// ISA layout: lanes 0-15 hold M=lane, K = ks*32 + {0..7, 16..23};
//             lanes 16-31 hold M=lane-16, K = ks*32 + {8..15, 24..31}.
__device__ __forceinline__ v16bf load_a_frag(const unsigned short* base, int stride,
                                             int mtile, int ks, int lane) {
    int m  = (mtile << 4) + (lane & 15);
    int kb = (ks << 5) + ((lane >> 4) << 3);
    const unsigned short* p = base + m * stride + kb;
    FragU u;
    u.h[0] = *(const ush8*)(p);
    u.h[1] = *(const ush8*)(p + 16);
    return u.v;
}

// B fragment from pre-packed weights: per (ntile,kstep,lane) 16 contiguous bf16.
__device__ __forceinline__ v16bf load_b_frag(const unsigned short* packed,
                                             int ntile, int ks, int ksteps, int lane) {
    const unsigned short* p = packed + ((((ntile * ksteps + ks) << 5) + lane) << 4);
    FragU u;
    u.h[0] = *(const ush8*)(p);
    u.h[1] = *(const ush8*)(p + 8);
    return u.v;
}

// ---------------------------------------------------------------------------
// K0: convert feats fp32 -> bf16 table (16 elements per thread).
// ---------------------------------------------------------------------------
__global__ __launch_bounds__(256) void cvt_feats_kernel(const float* __restrict__ src,
                                                        unsigned short* __restrict__ dst,
                                                        long long total) {
    long long base = ((long long)blockIdx.x * 256 + threadIdx.x) * 16;
    if (base >= total) return;
    const float4* s4 = (const float4*)(src + base);
    float4 x0 = s4[0], x1 = s4[1], x2 = s4[2], x3 = s4[3];
    uint4 p0, p1;
    p0.x = pk2bf(x0.x, x0.y); p0.y = pk2bf(x0.z, x0.w);
    p0.z = pk2bf(x1.x, x1.y); p0.w = pk2bf(x1.z, x1.w);
    p1.x = pk2bf(x2.x, x2.y); p1.y = pk2bf(x2.z, x2.w);
    p1.z = pk2bf(x3.x, x3.y); p1.w = pk2bf(x3.z, x3.w);
    *(uint4*)(dst + base)     = p0;
    *(uint4*)(dst + base + 8) = p1;
}

// ---------------------------------------------------------------------------
// K1: pack src[K][N] fp32 -> bf16 B-fragment order.
// dst index = ((ntile*ksteps + kstep)*32 + lane)*16 + j,
//   n = ntile*16 + (lane&15),  k = kstep*32 + (lane>>4)*8 + (j&7) + (j>>3)*16.
// ---------------------------------------------------------------------------
__global__ __launch_bounds__(256) void pack_b_kernel(const float* __restrict__ src,
                                                     unsigned short* __restrict__ dst,
                                                     int K, int N) {
    int tid = blockIdx.x * 256 + threadIdx.x;
    if (tid >= K * N) return;
    int j      = tid & 15;
    int lane   = (tid >> 4) & 31;
    int frag   = tid >> 9;
    int ksteps = K >> 5;
    int ntile  = frag / ksteps;
    int kstep  = frag - ntile * ksteps;
    int n = (ntile << 4) + (lane & 15);
    int k = (kstep << 5) + ((lane >> 4) << 3) + (j & 7) + ((j >> 3) << 4);
    dst[tid] = f2bf(src[k * N + n]);
}

// ---------------------------------------------------------------------------
// K2: one workgroup per query. Gather 32 bf16 neighbor rows (async -> LDS),
// gates f,i via WMMA, cell state c = mean_r( f*i*tanh(nbr) ).
// ---------------------------------------------------------------------------
__global__ __launch_bounds__(256) void gates_cell_kernel(
        const unsigned short* __restrict__ feats_bf, const int* __restrict__ nidx,
        const unsigned short* __restrict__ pWf, const unsigned short* __restrict__ pWi,
        const float* __restrict__ bf_, const float* __restrict__ bi_,
        float* __restrict__ c_out) {
    __shared__ __align__(16) unsigned short s_nbr[KNN * 136];   // bf16 rows, padded
    __shared__ __align__(16) float          s_gate[2][KNN * 132];

    const int q = blockIdx.x;
    const int t = threadIdx.x;

    // gather: 32 rows x 256B; 16B chunks, 512 chunks over 256 threads
#if USE_ASYNC_GATHER
    #pragma unroll
    for (int cc = 0; cc < 2; ++cc) {
        int ci  = t + (cc << 8);
        int r   = ci >> 4;
        int o16 = (ci & 15) << 3;                    // u16 offset in row
        int idx = nidx[q * KNN + r];
        const unsigned short* g = feats_bf + (size_t)idx * DD + o16;
        unsigned short*       l = s_nbr + r * 136 + o16;
        __builtin_amdgcn_global_load_async_to_lds_b128(
            (gptr_v4i)(unsigned short*)g, (lptr_v4i)l, 0, 0);
    }
    wait_async0();
#else
    {
        int r  = t >> 3;
        int c0 = (t & 7) << 4;
        int idx = nidx[q * KNN + r];
        const uint4* g = (const uint4*)(feats_bf + (size_t)idx * DD + c0);
        uint4 a0 = g[0], a1 = g[1];
        *(uint4*)(s_nbr + r * 136 + c0)     = a0;
        *(uint4*)(s_nbr + r * 136 + c0 + 8) = a1;
    }
#endif
    __syncthreads();

    const int lane = t & 31;
    const int wave = t >> 5;

    // 32 output tiles: gate(2) x mtile(2) x ntile(8); 4 per wave
    for (int tt = 0; tt < 4; ++tt) {
        int tau   = wave + (tt << 3);
        int g     = tau >> 4;
        int mtile = (tau >> 3) & 1;
        int ntile = tau & 7;
        const unsigned short* pw = g ? pWi : pWf;
        v8f acc = {};
        #pragma unroll
        for (int ks = 0; ks < 4; ++ks) {
            v16bf a = load_a_frag(s_nbr, 136, mtile, ks, lane);
            v16bf b = load_b_frag(pw, ntile, ks, 4, lane);
            acc = __builtin_amdgcn_wmma_f32_16x16x32_bf16(
                      false, a, false, b, (short)0, acc, false, false);
        }
        int n = (ntile << 4) + (lane & 15);
        float bias = g ? bi_[n] : bf_[n];
        float* gout = s_gate[g];
        #pragma unroll
        for (int r = 0; r < 8; ++r) {
            int m = (mtile << 4) + ((lane >> 4) << 3) + r;
            gout[m * 132 + n] = fast_sigmoid(acc[r] + bias);
        }
    }
    __syncthreads();

    // c[d] = (1/32) * sum_r f*i*tanh(x)
    if (t < DD) {
        float sum = 0.0f;
        #pragma unroll 4
        for (int r = 0; r < KNN; ++r) {
            float fv = s_gate[0][r * 132 + t];
            float iv = s_gate[1][r * 132 + t];
            float xv = bf2f(s_nbr[r * 136 + t]);
            sum += fv * iv * fast_tanh(xv);
        }
        c_out[(size_t)q * DD + t] = sum * (1.0f / KNN);
    }
}

// ---------------------------------------------------------------------------
// K3: 32 batch rows per workgroup. Fused:
//   o = sigmoid(self@Wo+bo); h = o*tanh(c); e = [self,h]@Wmv+bmv  (query & mv)
//   hidden = relu([eq,emv]@W1+b1); benefit = hidden@W2+b2
// LDS is a single arena with phase-overlaid buffers (peak 49.5 KB).
// ---------------------------------------------------------------------------
__global__ __launch_bounds__(256) void embed_mlp_kernel(
        const unsigned short* __restrict__ feats_bf,
        const int* __restrict__ qidx, const int* __restrict__ midx,
        const float* __restrict__ c_in,
        const unsigned short* __restrict__ pWo,  const float* __restrict__ bo_,
        const unsigned short* __restrict__ pWmv, const float* __restrict__ bmv_,
        const unsigned short* __restrict__ pW1,  const float* __restrict__ b1_,
        const float* __restrict__ W2, const float* __restrict__ b2_,
        float* __restrict__ out) {
    __shared__ __align__(16) unsigned char s_raw[50688];
    unsigned short* A2q   = (unsigned short*)(s_raw);            // [32][264] bf16 (self||h)
    unsigned short* A2m   = (unsigned short*)(s_raw + 16896);    // [32][264] bf16
    unsigned short* A3    = (unsigned short*)(s_raw + 33792);    // [32][264] bf16 (eq||emv)
    float*          tanhC = (float*)(s_raw + 33792);             // [32][132] overlays A3 (phase 1)
    float*          hidden= (float*)(s_raw);                     // [32][264] overlays A2q/A2m (phase 3)

    const int b0 = blockIdx.x << 5;
    const int t  = threadIdx.x;

    // gather self rows (bf16 copy) into A2 cols 0..127; tanh(c) scratch
    {
        int r  = t >> 3;
        int c0 = (t & 7) << 4;
        int qi = qidx[b0 + r];
        int mi = midx[b0 + r];
        const uint4* gq = (const uint4*)(feats_bf + (size_t)qi * DD + c0);
        const uint4* gm = (const uint4*)(feats_bf + (size_t)mi * DD + c0);
        uint4 q0 = gq[0], q1 = gq[1], m0 = gm[0], m1 = gm[1];
        *(uint4*)(A2q + r * 264 + c0)     = q0;
        *(uint4*)(A2q + r * 264 + c0 + 8) = q1;
        *(uint4*)(A2m + r * 264 + c0)     = m0;
        *(uint4*)(A2m + r * 264 + c0 + 8) = m1;
        const float4* sc = (const float4*)(c_in + (size_t)(b0 + r) * DD + c0);
        float* td = tanhC + r * 132 + c0;
        #pragma unroll
        for (int v = 0; v < 4; ++v) {
            float4 x = sc[v];
            float4 y;
            y.x = fast_tanh(x.x); y.y = fast_tanh(x.y);
            y.z = fast_tanh(x.z); y.w = fast_tanh(x.w);
            *(float4*)(td + 4 * v) = y;
        }
    }
    __syncthreads();

    const int lane = t & 31;
    const int wave = t >> 5;

    // --- GEMM1: o = sigmoid(self@Wo+bo); h = o*tanh(c) -> A2 cols 128..255 ---
    for (int tt = 0; tt < 4; ++tt) {
        int tau   = wave + (tt << 3);
        int srcm  = tau >> 4;
        int mtile = (tau >> 3) & 1;
        int ntile = tau & 7;
        unsigned short* A2 = srcm ? A2m : A2q;
        v8f acc = {};
        #pragma unroll
        for (int ks = 0; ks < 4; ++ks) {
            v16bf a = load_a_frag(A2, 264, mtile, ks, lane);
            v16bf b = load_b_frag(pWo, ntile, ks, 4, lane);
            acc = __builtin_amdgcn_wmma_f32_16x16x32_bf16(
                      false, a, false, b, (short)0, acc, false, false);
        }
        int n = (ntile << 4) + (lane & 15);
        float bias = bo_[n];
        #pragma unroll
        for (int r = 0; r < 8; ++r) {
            int m = (mtile << 4) + ((lane >> 4) << 3) + r;
            float o = fast_sigmoid(acc[r] + bias);
            A2[m * 264 + 128 + n] = f2bf(o * tanhC[m * 132 + n]);
        }
    }
    __syncthreads();

    // --- GEMM2: e = [self,h]@Wmv + bmv  (K=256) -> A3 (q cols 0..127, mv 128..255)
    for (int tt = 0; tt < 4; ++tt) {
        int tau   = wave + (tt << 3);
        int srcm  = tau >> 4;
        int mtile = (tau >> 3) & 1;
        int ntile = tau & 7;
        unsigned short* A2 = srcm ? A2m : A2q;
        v8f acc = {};
        #pragma unroll
        for (int ks = 0; ks < 8; ++ks) {
            v16bf a = load_a_frag(A2, 264, mtile, ks, lane);
            v16bf b = load_b_frag(pWmv, ntile, ks, 8, lane);
            acc = __builtin_amdgcn_wmma_f32_16x16x32_bf16(
                      false, a, false, b, (short)0, acc, false, false);
        }
        int n = (ntile << 4) + (lane & 15);
        float bias = bmv_[n];
        #pragma unroll
        for (int r = 0; r < 8; ++r) {
            int m = (mtile << 4) + ((lane >> 4) << 3) + r;
            A3[m * 264 + (srcm << 7) + n] = f2bf(acc[r] + bias);
        }
    }
    __syncthreads();

    // --- GEMM3: hidden = relu([eq,emv]@W1 + b1)  (K=256, N=256) ---
    for (int tt = 0; tt < 4; ++tt) {
        int tau   = wave + (tt << 3);
        int mtile = tau >> 4;
        int ntile = tau & 15;
        v8f acc = {};
        #pragma unroll
        for (int ks = 0; ks < 8; ++ks) {
            v16bf a = load_a_frag(A3, 264, mtile, ks, lane);
            v16bf b = load_b_frag(pW1, ntile, ks, 8, lane);
            acc = __builtin_amdgcn_wmma_f32_16x16x32_bf16(
                      false, a, false, b, (short)0, acc, false, false);
        }
        int n = (ntile << 4) + (lane & 15);
        float bias = b1_[n];
        #pragma unroll
        for (int r = 0; r < 8; ++r) {
            int m = (mtile << 4) + ((lane >> 4) << 3) + r;
            hidden[m * 264 + n] = fmaxf(acc[r] + bias, 0.0f);
        }
    }
    __syncthreads();

    // --- final dot with W2 (H=256 -> 1): 8 threads per row + shfl reduce ---
    {
        int row = t >> 3;
        int seg = t & 7;
        float p = 0.0f;
        #pragma unroll 8
        for (int j = 0; j < 32; ++j) {
            int n = (seg << 5) + j;
            p += hidden[row * 264 + n] * W2[n];
        }
        p += __shfl_xor(p, 1, 8);
        p += __shfl_xor(p, 2, 8);
        p += __shfl_xor(p, 4, 8);
        if (seg == 0) out[b0 + row] = p + b2_[0];
    }
}

// ---------------------------------------------------------------------------
extern "C" void kernel_launch(void* const* d_in, const int* in_sizes, int n_in,
                              void* d_out, int out_size, void* d_ws, size_t ws_size,
                              hipStream_t stream) {
    const float* feats        = (const float*)d_in[0];
    const int*   query_idx    = (const int*)d_in[1];
    const int*   mv_idx       = (const int*)d_in[2];
    const int*   neighbor_idx = (const int*)d_in[3];
    const float* Wf  = (const float*)d_in[4];
    const float* bf_ = (const float*)d_in[5];
    const float* Wi  = (const float*)d_in[6];
    const float* bi_ = (const float*)d_in[7];
    const float* Wo  = (const float*)d_in[8];
    const float* bo_ = (const float*)d_in[9];
    const float* Wmv = (const float*)d_in[10];
    const float* bmv_= (const float*)d_in[11];
    const float* W1  = (const float*)d_in[12];
    const float* b1_ = (const float*)d_in[13];
    const float* W2  = (const float*)d_in[14];
    const float* b2_ = (const float*)d_in[15];

    unsigned char* ws = (unsigned char*)d_ws;
    unsigned short* pWf  = (unsigned short*)(ws);             //  32 KB
    unsigned short* pWi  = (unsigned short*)(ws + 32768);     //  32 KB
    unsigned short* pWo  = (unsigned short*)(ws + 65536);     //  32 KB
    unsigned short* pWmv = (unsigned short*)(ws + 98304);     //  64 KB
    unsigned short* pW1  = (unsigned short*)(ws + 163840);    // 128 KB
    float*          c_ws = (float*)(ws + 294912);             //   8 MB (B*D f32)
    unsigned short* f_bf = (unsigned short*)(ws + 8683520);   // 128 MB bf16 feats

    const int       B  = in_sizes[1];            // 16384
    const long long FE = (long long)in_sizes[0]; // N*D = 64M

    cvt_feats_kernel<<<(unsigned)((FE / 16 + 255) / 256), 256, 0, stream>>>(feats, f_bf, FE);

    pack_b_kernel<<<64,  256, 0, stream>>>(Wf,  pWf,  128, 128);
    pack_b_kernel<<<64,  256, 0, stream>>>(Wi,  pWi,  128, 128);
    pack_b_kernel<<<64,  256, 0, stream>>>(Wo,  pWo,  128, 128);
    pack_b_kernel<<<128, 256, 0, stream>>>(Wmv, pWmv, 256, 128);
    pack_b_kernel<<<256, 256, 0, stream>>>(W1,  pW1,  256, 256);

    gates_cell_kernel<<<B, 256, 0, stream>>>(f_bf, neighbor_idx, pWf, pWi,
                                             bf_, bi_, c_ws);

    embed_mlp_kernel<<<B / 32, 256, 0, stream>>>(f_bf, query_idx, mv_idx, c_ws,
                                                 pWo, bo_, pWmv, bmv_, pW1, b1_,
                                                 W2, b2_, (float*)d_out);
}
